// MLMPS_69672959476118
// MI455X (gfx1250) — matrix operand order repaired and checked by reading, not compiled, not run on hardware
//
#include <hip/hip_runtime.h>
#include <hip/hip_bf16.h>

// MI455X (gfx1250) implementation.
// Layer-1 bottleneck: streaming Am_1 (160 MB) once -> ~7us HBM floor @23.3TB/s.
// Site-parallel WMMA GEMMs produce per-batch 32x32 matrices; a cheap wave32
// register-resident scan (shfl broadcast + shfl_xor max-rescale) consumes them.
// K and k-step count are template params so every fragment index is a
// compile-time constant (no scratch spills / v_movrels around v_wmma).

typedef _Float16 h16;
typedef _Float16 v16h __attribute__((ext_vector_type(16)));
typedef _Float16 h8   __attribute__((ext_vector_type(8)));
typedef float    v8f  __attribute__((ext_vector_type(8)));

#define PI_F 3.14159265358979323846f

// ---------------- trig embedding (per-site norm is 1: scaled by 1/sqrt(9C)) -------
__global__ void k_embed(const float* __restrict__ img, int H, int W, int C,
                        h16* __restrict__ Sh, float* __restrict__ Sedge,
                        float* __restrict__ Sfull) {
  int N = (H / 3) * (W / 3);
  int F27 = 9 * C;
  int F = 2 * F27;
  int t = blockIdx.x * blockDim.x + threadIdx.x;
  if (t >= 64 * N) return;
  int b = t / N, site = t % N;
  int W3 = W / 3;
  int h0 = site / W3, w0 = site % W3;
  float inv = rsqrtf((float)F27);
  h16* dst = Sh + ((size_t)b * N + site) * 64;
  for (int f27 = 0; f27 < F27; ++f27) {
    int dhdw = f27 / C, c = f27 % C;
    int dh = dhdw / 3, dw = dhdw % 3;
    float p = img[(((size_t)b * H + h0 * 3 + dh) * W + (w0 * 3 + dw)) * C + c];
    float ang = 0.5f * PI_F * p;
    float co = cosf(ang) * inv, si = sinf(ang) * inv;
    dst[2 * f27]     = (h16)co;
    dst[2 * f27 + 1] = (h16)si;
    if (Sedge) {
      if (site == 0)     { Sedge[(size_t)b*2*F + 2*f27] = co;     Sedge[(size_t)b*2*F + 2*f27 + 1] = si; }
      if (site == N - 1) { Sedge[(size_t)b*2*F + F + 2*f27] = co; Sedge[(size_t)b*2*F + F + 2*f27 + 1] = si; }
    }
    if (Sfull) {
      Sfull[((size_t)b*N + site)*F + 2*f27]     = co;
      Sfull[((size_t)b*N + site)*F + 2*f27 + 1] = si;
    }
  }
  for (int k = F; k < 64; ++k) dst[k] = (h16)0.0f;
}

// ------------- per-site GEMM: P[site][b][x][y] = sum_f Am[x,y,f]*s[b,f] ----
// C(1024x64) = A(1024xKpad) * B(Kpadx64) via v_wmma_f32_16x16x32_f16.
// KDIM = true K (54 or 18), NK = number of K=32 steps (2 or 1).
template <int KDIM, int NK>
__global__ __launch_bounds__(256) void k_site_gemm(
    const float* __restrict__ Am, const h16* __restrict__ S,
    float* __restrict__ P, int siteBase, int nSitesS) {
  __shared__ h16 smA[256 * 64];   // 32 KB: one 256-row slice of A, f16, K padded
  __shared__ h16 smB[64 * 64];    //  8 KB: s row (site+1) for all 64 batches
  const int tid = threadIdx.x;
  const int wave = tid >> 5, lane = tid & 31;
  const int site = siteBase + blockIdx.x;
  const size_t Abase = (size_t)site * 1024 * KDIM;

  for (int idx = tid; idx < 64 * 64; idx += 256) {
    int b = idx >> 6, k = idx & 63;
    smB[idx] = S[((size_t)b * nSitesS + site + 1) * 64 + k];
  }
  __syncthreads();

  // B fragments (shared across all M tiles): lane holds n=lane&15,
  // K = 16*(lane>>4) + [0..15] (+32*ks) -> 32 contiguous bytes in LDS.
  v16h bfr[4][NK];
  {
    int n = lane & 15, g = lane >> 4;
#pragma unroll
    for (int nt = 0; nt < 4; ++nt)
#pragma unroll
      for (int ks = 0; ks < NK; ++ks) {
        const h8* pb = (const h8*)&smB[(nt * 16 + n) * 64 + ks * 32 + 16 * g];
        bfr[nt][ks] = __builtin_shufflevector(pb[0], pb[1],
            0,1,2,3,4,5,6,7,8,9,10,11,12,13,14,15);
      }
  }

  for (int slice = 0; slice < 4; ++slice) {
    __syncthreads();
    for (int idx = tid; idx < 256 * 64; idx += 256) {
      int row = idx >> 6, k = idx & 63;
      float v = (k < KDIM) ? Am[Abase + (size_t)(slice * 256 + row) * KDIM + k] : 0.0f;
      smA[idx] = (h16)v;
    }
    if (slice < 3)  // global_prefetch_b8 for the next slice
      __builtin_prefetch(Am + Abase + (size_t)((slice + 1) * 256 + tid) * KDIM, 0, 1);
    __syncthreads();

#pragma unroll
    for (int tpair = 0; tpair < 2; ++tpair) {
      int mloc = wave * 2 + tpair;
      int arow = mloc * 16 + (lane & 15);
      int g = lane >> 4;
      // A fragment: lane m=lane&15, K = 16*(lane>>4) + [0..15].
      v16h afr[NK];
#pragma unroll
      for (int ks = 0; ks < NK; ++ks) {
        const h8* pa0 = (const h8*)&smA[arow * 64 + ks * 32 + 16 * g];
        const h8* pa1 = (const h8*)&smA[arow * 64 + ks * 32 + 16 * g + 8];
        afr[ks] = __builtin_shufflevector(pa0[0], pa1[0],
            0,1,2,3,4,5,6,7,8,9,10,11,12,13,14,15);
      }
#pragma unroll
      for (int nt = 0; nt < 4; ++nt) {
        v8f acc = {};
#pragma unroll
        for (int ks = 0; ks < NK; ++ks)
          acc = __builtin_amdgcn_wmma_f32_16x16x32_f16(
              false, afr[ks], false, bfr[nt][ks], (short)0, acc, false, false);
        // C/D layout: col n=lane&15, rows M = v + 8*(lane>>4) -> 8 contiguous floats
        int ncol = nt * 16 + (lane & 15);
        int xy = slice * 256 + mloc * 16 + 8 * g;
        float* pd = P + ((size_t)blockIdx.x * 64 + ncol) * 1024 + xy;
        *(float4*)(pd)     = make_float4(acc[0], acc[1], acc[2], acc[3]);
        *(float4*)(pd + 4) = make_float4(acc[4], acc[5], acc[6], acc[7]);
      }
    }
  }
}

// ------- reverse scan over a chunk: r <- M_j r, per-batch max-rescale ------
__global__ __launch_bounds__(1024) void k_chain(
    const float* __restrict__ P, int nsites, float* __restrict__ r,
    float* __restrict__ lg, int trackLg) {
  int tid = threadIdx.x;
  int wave = tid >> 5, lane = tid & 31;
  int b0 = wave, b1 = wave + 32;
  float rA = r[b0 * 32 + lane];
  float rB = r[b1 * 32 + lane];
  float lgA = 0.f, lgB = 0.f;
  for (int j = nsites - 1; j >= 0; --j) {
    const float* pA = P + ((size_t)j * 64 + b0) * 1024 + lane * 32;
    const float* pB = P + ((size_t)j * 64 + b1) * 1024 + lane * 32;
    float aA = 0.f, aB = 0.f;
#pragma unroll
    for (int y4 = 0; y4 < 8; ++y4) {
      float4 mA = *(const float4*)(pA + 4 * y4);
      float4 mB = *(const float4*)(pB + 4 * y4);
      aA += mA.x * __shfl(rA, 4*y4+0) + mA.y * __shfl(rA, 4*y4+1)
          + mA.z * __shfl(rA, 4*y4+2) + mA.w * __shfl(rA, 4*y4+3);
      aB += mB.x * __shfl(rB, 4*y4+0) + mB.y * __shfl(rB, 4*y4+1)
          + mB.z * __shfl(rB, 4*y4+2) + mB.w * __shfl(rB, 4*y4+3);
    }
    float mA2 = fabsf(aA), mB2 = fabsf(aB);
#pragma unroll
    for (int off = 16; off >= 1; off >>= 1) {
      mA2 = fmaxf(mA2, __shfl_xor(mA2, off));
      mB2 = fmaxf(mB2, __shfl_xor(mB2, off));
    }
    float scA = mA2 + 1e-30f, scB = mB2 + 1e-30f;
    rA = aA / scA; rB = aB / scB;
    if (trackLg) { lgA += logf(scA); lgB += logf(scB); }
  }
  r[b0 * 32 + lane] = rA;
  r[b1 * 32 + lane] = rB;
  if (trackLg && lane == 0) { lg[b0] += lgA; lg[b1] += lgB; }
}

// r0[b][x] = sum_f s_last[b][f] * Al[x][f] ; lg = 0
__global__ void k_init_r(const float* __restrict__ Sedge, const float* __restrict__ Al,
                         int F, float* __restrict__ r, float* __restrict__ lg) {
  int b = blockIdx.x, x = threadIdx.x;
  float acc = 0.f;
  for (int f = 0; f < F; ++f) acc += Sedge[((size_t)b * 2 + 1) * F + f] * Al[x * F + f];
  r[b * 32 + x] = acc;
  if (x == 0) lg[b] = 0.f;
}

// v[b][o] = sum_x r[b][x] * sum_f s0[b][f] * A0[x][f][o]
__global__ void k_edge_v(const float* __restrict__ Sedge, const float* __restrict__ A0,
                         const float* __restrict__ r, int F, int O, float* __restrict__ v) {
  int o = blockIdx.x, b = threadIdx.x;
  float acc = 0.f;
  for (int x = 0; x < 32; ++x) {
    float p = 0.f;
    for (int f = 0; f < F; ++f) p += Sedge[(size_t)b * 2 * F + f] * A0[((size_t)x * F + f) * O + o];
    acc += r[b * 32 + x] * p;
  }
  v[(size_t)b * O + o] = acc;
}

__global__ void k_rownorm(const float* __restrict__ v, int O, float* __restrict__ rn) {
  __shared__ float red[256];
  int b = blockIdx.x;
  float s = 0.f;
  for (int o = threadIdx.x; o < O; o += 256) { float t = v[(size_t)b * O + o]; s += t * t; }
  red[threadIdx.x] = s; __syncthreads();
  for (int off = 128; off >= 1; off >>= 1) {
    if (threadIdx.x < off) red[threadIdx.x] += red[threadIdx.x + off];
    __syncthreads();
  }
  if (threadIdx.x == 0) rn[b] = sqrtf(red[0]);
}

// training-mode BatchNorm over batch axis (biased variance), input v/rn
__global__ void k_bn(const float* __restrict__ v, const float* __restrict__ rn,
                     const float* __restrict__ g, const float* __restrict__ bb,
                     int O, float* __restrict__ out) {
  int o = blockIdx.x * blockDim.x + threadIdx.x;
  if (o >= O) return;
  float m1 = 0.f, m2 = 0.f;
  for (int b = 0; b < 64; ++b) { float t = v[(size_t)b * O + o] / rn[b]; m1 += t; m2 += t * t; }
  float mu = m1 * (1.f / 64.f);
  float var = m2 * (1.f / 64.f) - mu * mu;
  float is = rsqrtf(var + 1e-5f);
  for (int b = 0; b < 64; ++b) {
    float t = v[(size_t)b * O + o] / rn[b];
    out[(size_t)b * O + o] = g[o] * (t - mu) * is + bb[o];
  }
}

// ---------------- layer 3: full chain, one wave per batch ------------------
__global__ void k_l3_apply(const float* __restrict__ S, const float* __restrict__ A0,
                           const float* __restrict__ Am, const float* __restrict__ Al,
                           float* __restrict__ v3, float* __restrict__ lg3) {
  int b = blockIdx.x, lane = threadIdx.x;  // 32 threads
  float rl = 0.f;
  for (int f = 0; f < 18; ++f) rl += S[((size_t)b * 9 + 8) * 18 + f] * Al[lane * 18 + f];
  float lg = 0.f;
  for (int j = 6; j >= 0; --j) {
    float sv[18];
#pragma unroll
    for (int f = 0; f < 18; ++f) sv[f] = S[((size_t)b * 9 + j + 1) * 18 + f];
    float acc = 0.f;
    for (int y = 0; y < 32; ++y) {
      float w = 0.f;
      const float* a = Am + (((size_t)(j * 32 + lane)) * 32 + y) * 18;
#pragma unroll
      for (int f = 0; f < 18; ++f) w += a[f] * sv[f];
      acc += w * __shfl(rl, y);
    }
    float m = fabsf(acc);
    for (int off = 16; off >= 1; off >>= 1) m = fmaxf(m, __shfl_xor(m, off));
    float sc = m + 1e-30f;
    rl = acc / sc;
    lg += logf(sc);
  }
  float s0[18];
#pragma unroll
  for (int f = 0; f < 18; ++f) s0[f] = S[((size_t)b * 9) * 18 + f];
  for (int o = 0; o < 10; ++o) {
    float wv = 0.f;
#pragma unroll
    for (int f = 0; f < 18; ++f) wv += s0[f] * A0[((size_t)lane * 18 + f) * 10 + o];
    float c = rl * wv;
    for (int off = 16; off >= 1; off >>= 1) c += __shfl_xor(c, off);
    if (lane == 0) v3[b * 10 + o] = c;
  }
  if (lane == 0) lg3[b] = lg;
}

// Frobenius norm of the layer-3 MPO via transfer matrices
__global__ __launch_bounds__(1024) void k_chain_norm(
    const float* __restrict__ A0, const float* __restrict__ Al,
    const float* __restrict__ Am, float* __restrict__ Wbuf,
    float* __restrict__ nmbuf) {
  __shared__ float T[32 * 32];
  __shared__ float red[32];
  int tid = threadIdx.x;
  int a = tid >> 5, cc = tid & 31;
  {
    float s = 0.f;
    for (int f = 0; f < 18; ++f)
      for (int o = 0; o < 10; ++o)
        s += A0[((size_t)a * 18 + f) * 10 + o] * A0[((size_t)cc * 18 + f) * 10 + o];
    T[tid] = s;
  }
  __syncthreads();
  float lgs = 0.f;
  for (int j = 0; j < 7; ++j) {
    // W[y,a,f] = sum_x T[x,y] A[x,a,f]
    for (int idx = tid; idx < 32 * 32 * 18; idx += 1024) {
      int y = idx / 576, rem = idx % 576, aa = rem / 18, f = rem % 18;
      float s = 0.f;
      for (int x = 0; x < 32; ++x)
        s += T[x * 32 + y] * Am[(((size_t)(j * 32 + x)) * 32 + aa) * 18 + f];
      Wbuf[idx] = s;
    }
    __syncthreads();
    // T'[a,b] = sum_{y,f} W[y,a,f] A[y,b,f]
    float t = 0.f;
    for (int y = 0; y < 32; ++y) {
      const float* wp = Wbuf + ((size_t)y * 32 + a) * 18;
      const float* mp = Am + (((size_t)(j * 32 + y)) * 32 + cc) * 18;
      for (int f = 0; f < 18; ++f) t += wp[f] * mp[f];
    }
    float mv = fabsf(t);
    for (int off = 16; off >= 1; off >>= 1) mv = fmaxf(mv, __shfl_xor(mv, off));
    if ((tid & 31) == 0) red[tid >> 5] = mv;
    __syncthreads();
    if (tid < 32) {
      float mm = red[tid];
      for (int off = 16; off >= 1; off >>= 1) mm = fmaxf(mm, __shfl_xor(mm, off));
      if (tid == 0) red[0] = mm;
    }
    __syncthreads();
    float sc = red[0] + 1e-30f;
    T[tid] = t / sc;
    lgs += logf(sc);
    __syncthreads();
  }
  float al = 0.f;
  for (int f = 0; f < 18; ++f) al += Al[a * 18 + f] * Al[cc * 18 + f];
  float s = T[tid] * al;
  for (int off = 16; off >= 1; off >>= 1) s += __shfl_xor(s, off);
  if ((tid & 31) == 0) red[tid >> 5] = s;
  __syncthreads();
  if (tid < 32) {
    float ss = red[tid];
    for (int off = 16; off >= 1; off >>= 1) ss += __shfl_xor(ss, off);
    if (tid == 0) { nmbuf[0] = sqrtf(ss); nmbuf[1] = 0.5f * lgs; }
  }
}

__global__ void k_final(const float* __restrict__ v3, const float* __restrict__ lg3,
                        const float* __restrict__ nmbuf, float* __restrict__ out) {
  int t = blockIdx.x * blockDim.x + threadIdx.x;
  if (t >= 640) return;
  int b = t / 10;
  out[t] = v3[t] * expf(lg3[b] - nmbuf[1]) / nmbuf[0];
}

// ----------------------------------------------------------------------------
extern "C" void kernel_launch(void* const* d_in, const int* in_sizes, int n_in,
                              void* d_out, int out_size, void* d_ws, size_t ws_size,
                              hipStream_t stream) {
  const float* x    = (const float*)d_in[0];
  const float* A0_1 = (const float*)d_in[1];
  const float* Am_1 = (const float*)d_in[2];
  const float* Al_1 = (const float*)d_in[3];
  const float* g1   = (const float*)d_in[4];
  const float* b1   = (const float*)d_in[5];
  const float* A0_2 = (const float*)d_in[6];
  const float* Am_2 = (const float*)d_in[7];
  const float* Al_2 = (const float*)d_in[8];
  const float* g2   = (const float*)d_in[9];
  const float* b2   = (const float*)d_in[10];
  const float* A0_3 = (const float*)d_in[11];
  const float* Am_3 = (const float*)d_in[12];
  const float* Al_3 = (const float*)d_in[13];
  float* out = (float*)d_out;
  (void)in_sizes; (void)n_in; (void)out_size; (void)ws_size;

  char* w = (char*)d_ws;
  auto alloc = [&](size_t bytes) {
    void* p = (void*)w;
    w += (bytes + 255) & ~(size_t)255;
    return p;
  };
  h16*   s1h  = (h16*)  alloc((size_t)64 * 729 * 64 * 2);
  float* se1  = (float*)alloc((size_t)64 * 2 * 54 * 4);
  h16*   s2h  = (h16*)  alloc((size_t)64 * 81 * 64 * 2);
  float* se2  = (float*)alloc((size_t)64 * 2 * 18 * 4);
  h16*   s3h  = (h16*)  alloc((size_t)64 * 9 * 64 * 2);
  float* se3  = (float*)alloc((size_t)64 * 2 * 18 * 4);
  float* s3f  = (float*)alloc((size_t)64 * 9 * 18 * 4);
  float* Pbuf = (float*)alloc((size_t)64 * 64 * 1024 * 4);   // 16.8 MB chunk
  float* rbuf = (float*)alloc((size_t)64 * 32 * 4);
  float* lgb  = (float*)alloc((size_t)64 * 4);
  float* v1   = (float*)alloc((size_t)64 * 729 * 4);
  float* img1 = (float*)alloc((size_t)64 * 729 * 4);
  float* v2   = (float*)alloc((size_t)64 * 81 * 4);
  float* img2 = (float*)alloc((size_t)64 * 81 * 4);
  float* rn   = (float*)alloc((size_t)64 * 4);
  float* v3   = (float*)alloc((size_t)64 * 10 * 4);
  float* lg3  = (float*)alloc((size_t)64 * 4);
  float* Wb   = (float*)alloc((size_t)32 * 32 * 18 * 4);
  float* nmb  = (float*)alloc((size_t)2 * 4);

  const int CH = 64;

  // ---- layer 1: 729 sites, F=54, 727 middle sites ----
  k_embed<<<(64 * 729 + 255) / 256, 256, 0, stream>>>(x, 81, 81, 3, s1h, se1, nullptr);
  k_init_r<<<64, 32, 0, stream>>>(se1, Al_1, 54, rbuf, lgb);
  {
    int nmid = 727, nch = (nmid + CH - 1) / CH;
    for (int c = nch - 1; c >= 0; --c) {
      int s0 = c * CH, cnt = (nmid - s0 < CH) ? (nmid - s0) : CH;
      k_site_gemm<54, 2><<<cnt, 256, 0, stream>>>(Am_1, s1h, Pbuf, s0, 729);
      k_chain<<<1, 1024, 0, stream>>>(Pbuf, cnt, rbuf, nullptr, 0);
    }
  }
  k_edge_v<<<729, 64, 0, stream>>>(se1, A0_1, rbuf, 54, 729, v1);
  k_rownorm<<<64, 256, 0, stream>>>(v1, 729, rn);
  k_bn<<<(729 + 255) / 256, 256, 0, stream>>>(v1, rn, g1, b1, 729, img1);

  // ---- layer 2: 81 sites, F=18, 79 middle sites ----
  k_embed<<<(64 * 81 + 255) / 256, 256, 0, stream>>>(img1, 27, 27, 1, s2h, se2, nullptr);
  k_init_r<<<64, 32, 0, stream>>>(se2, Al_2, 18, rbuf, lgb);
  {
    int nmid = 79, nch = (nmid + CH - 1) / CH;
    for (int c = nch - 1; c >= 0; --c) {
      int s0 = c * CH, cnt = (nmid - s0 < CH) ? (nmid - s0) : CH;
      k_site_gemm<18, 1><<<cnt, 256, 0, stream>>>(Am_2, s2h, Pbuf, s0, 81);
      k_chain<<<1, 1024, 0, stream>>>(Pbuf, cnt, rbuf, nullptr, 0);
    }
  }
  k_edge_v<<<81, 64, 0, stream>>>(se2, A0_2, rbuf, 18, 81, v2);
  k_rownorm<<<64, 256, 0, stream>>>(v2, 81, rn);
  k_bn<<<1, 256, 0, stream>>>(v2, rn, g2, b2, 81, img2);

  // ---- layer 3: 9 sites, F=18, out=10 (needs log-scale + model norm) ----
  k_embed<<<(64 * 9 + 255) / 256, 256, 0, stream>>>(img2, 9, 9, 1, s3h, se3, s3f);
  k_chain_norm<<<1, 1024, 0, stream>>>(A0_3, Al_3, Am_3, Wb, nmb);
  k_l3_apply<<<64, 32, 0, stream>>>(s3f, A0_3, Am_3, Al_3, v3, lg3);
  k_final<<<3, 256, 0, stream>>>(v3, lg3, nmb, out);
}
